// LLaDABlock_46188078301746
// MI455X (gfx1250) — matrix operand order repaired and checked by reading, not compiled, not run on hardware
//
#include <hip/hip_runtime.h>
#include <hip/hip_bf16.h>

typedef __bf16 bf16_t;
typedef __attribute__((ext_vector_type(16))) __bf16 v16bf;
typedef __attribute__((ext_vector_type(8)))  __bf16 v8bf;
typedef __attribute__((ext_vector_type(8)))  float  v8f;
typedef __attribute__((ext_vector_type(4)))  unsigned int v4u;
typedef __attribute__((ext_vector_type(8)))  int v8i;
typedef __attribute__((ext_vector_type(4)))  int v4i;

#define T_TOK 4096
#define DMODEL 1024
#define NEXP   8
#define FFN    4096
#define NHEADS 16
#define HDIM   64
#define SEQ    1024

// ---------------- WMMA helpers (CDNA5 gfx1250, wave32) ----------------

__device__ __forceinline__ v8f wmma_bf16f32(v16bf a, v16bf b, v8f c) {
  // D = A(16x32 bf16) * B(32x16 bf16) + C(16x16 f32)
  return __builtin_amdgcn_wmma_f32_16x16x32_bf16(false, a, false, b, (short)0, c,
                                                 false, false);
}

__device__ __forceinline__ v16bf joinv(v8bf lo, v8bf hi) {
  return __builtin_shufflevector(lo, hi, 0, 1, 2, 3, 4, 5, 6, 7, 8, 9, 10, 11, 12, 13, 14, 15);
}

// A-frag 16x32: lane holds row m0+(lane&15); K octets [koff,koff+8) and [koff+16,koff+24), koff=(lane>>4)*8
__device__ __forceinline__ v16bf load_A_frag(const bf16_t* A, int lda, int m0, int k0, int lane) {
  const bf16_t* p = A + (size_t)(m0 + (lane & 15)) * lda + k0 + ((lane >> 4) << 3);
  v8bf lo = *(const v8bf*)p;
  v8bf hi = *(const v8bf*)(p + 16);
  return joinv(lo, hi);
}

// A-frag from an already-offset row pointer (gather GEMMs): arow points at row base + koff
__device__ __forceinline__ v16bf load_A_row(const bf16_t* arow, int k0) {
  v8bf lo = *(const v8bf*)(arow + k0);
  v8bf hi = *(const v8bf*)(arow + k0 + 16);
  return joinv(lo, hi);
}

// B-frag 32x16 sourced from TRANSPOSED B (BT[N][K], row-major):
// K octets [koff,koff+8) and [koff+16,koff+24), koff=(lane>>4)*8, from BT[n0+(lane&15)][k0 + ...]
__device__ __forceinline__ v16bf load_BT_frag(const bf16_t* BT, int ldk, int n0, int k0, int lane) {
  const bf16_t* p = BT + (size_t)(n0 + (lane & 15)) * ldk + k0 + ((lane >> 4) << 3);
  v8bf lo = *(const v8bf*)p;
  v8bf hi = *(const v8bf*)(p + 16);
  return joinv(lo, hi);
}

__device__ __forceinline__ v16bf load_A_lds(const bf16_t* P, int ld, int k0, int lane) {
  const bf16_t* p = P + (lane & 15) * ld + k0 + ((lane >> 4) << 3);
  v8bf lo = *(const v8bf*)p;
  v8bf hi = *(const v8bf*)(p + 16);
  return joinv(lo, hi);
}

__device__ __forceinline__ float gelu_f(float x) {
  const float c = 0.7978845608028654f;
  return 0.5f * x * (1.f + tanhf(c * (x + 0.044715f * x * x * x)));
}

// ---------------- TDM: 2-D tile DMA global -> LDS (CDNA5 Tensor Data Mover) ----------------
// D# per ISA 8.3/8.4: count=1, type=2 (image), data_size=2B, pad_enable for bank-conflict-free LDS.

__device__ __forceinline__ void tdm_load_2d(unsigned lds_off, const bf16_t* gaddr,
                                            int tile_k, int rows, int ldk_elems) {
  unsigned long long ga = (unsigned long long)(uintptr_t)gaddr;
  v4u g0;
  g0[0] = 1u;                                                // count=1, user mode
  g0[1] = lds_off;                                           // LDS byte address
  g0[2] = (unsigned)(ga & 0xffffffffu);                      // global_addr[31:0]
  g0[3] = (unsigned)((ga >> 32) & 0x1ffffffu) | (2u << 30);  // global_addr[56:32] | type=2
  unsigned td0 = (unsigned)tile_k;   // tensor dims == tile dims: tile always in-bounds
  unsigned td1 = (unsigned)rows;
  unsigned long long s0 = (unsigned long long)(unsigned)ldk_elems;
  v8i g1;
  // data_size=1 (2B) | pad_enable | pad_interval=code4 (32 dwords) | pad_amount=code3 (4 dwords)
  g1[0] = (int)((1u << 16) | (1u << 20) | (4u << 22) | (3u << 25));
  g1[1] = (int)((td0 & 0xffffu) << 16);                       // tensor_dim0[15:0] @ bits63:48
  g1[2] = (int)((td0 >> 16) | ((td1 & 0xffffu) << 16));       // td0[31:16] | td1[15:0]
  g1[3] = (int)((td1 >> 16) | ((unsigned)tile_k << 16));      // td1[31:16] | tile_dim0
  g1[4] = (int)(unsigned)rows;                                // tile_dim1 (tile_dim2 = 0)
  g1[5] = (int)(unsigned)(s0 & 0xffffffffu);                  // tensor_dim0_stride lo
  g1[6] = (int)(unsigned)(s0 >> 32);                          // stride hi | dim1 stride lo = 0
  g1[7] = 0;
  v4i z4 = {0, 0, 0, 0};
  v8i z8 = {0, 0, 0, 0, 0, 0, 0, 0};
  __builtin_amdgcn_tensor_load_to_lds(g0, g1, z4, z4, z8, 0);
}

// TDM-fed, double-buffered WMMA mainloop: wave tile 16m x 128n, block tile 128m x 128n.
// B tile (128 rows x 64 k bf16) staged to LDS by the TDM; LDS rows padded to 144 B.
#define TDM_KCH  64
#define TDM_ROWB (TDM_KCH * 2 + 16)       // 144 B LDS row stride (pad 4 dwords / 32 dwords)
#define TDM_BUFB (128 * TDM_ROWB)         // 18432 B per buffer

template <int K>
__device__ __forceinline__ void gemm_mainloop_tdm(const bf16_t* arow, const bf16_t* Wtile,
                                                  int ldk, int lane, int wave,
                                                  char* sbuf, unsigned sbase, v8f c[8]) {
  const int NCH = K / TDM_KCH;
  if (wave == 0) {
    tdm_load_2d(sbase, Wtile, TDM_KCH, 128, ldk);
    tdm_load_2d(sbase + TDM_BUFB, Wtile + TDM_KCH, TDM_KCH, 128, ldk);
  }
  for (int ch = 0; ch < NCH; ++ch) {
    if (wave == 0) {
      if (ch + 1 < NCH)
        __builtin_amdgcn_s_wait_tensorcnt(1);
      else
        __builtin_amdgcn_s_wait_tensorcnt(0);
    }
    __syncthreads();  // buffer (ch&1) now holds chunk ch
    const char* buf = sbuf + (ch & 1) * TDM_BUFB;
    if (ch * TDM_KCH + 256 < K) __builtin_prefetch((const void*)(arow + ch * TDM_KCH + 256), 0, 3);
#pragma unroll
    for (int kk = 0; kk < TDM_KCH; kk += 32) {
      v16bf a = load_A_row(arow, ch * TDM_KCH + kk);
#pragma unroll
      for (int j = 0; j < 8; ++j) {
        const bf16_t* p =
            (const bf16_t*)(buf + (j * 16 + (lane & 15)) * TDM_ROWB) + kk + ((lane >> 4) << 3);
        v16bf b = joinv(*(const v8bf*)p, *(const v8bf*)(p + 16));
        c[j] = wmma_bf16f32(a, b, c[j]);
      }
    }
    __syncthreads();  // all reads of buffer (ch&1) done before TDM overwrites it
    if (wave == 0 && ch + 2 < NCH)
      tdm_load_2d(sbase + (ch & 1) * TDM_BUFB, Wtile + (size_t)(ch + 2) * TDM_KCH, TDM_KCH, 128, ldk);
  }
}

// Register software-pipelined mainloop (direct global loads) for the small L2-hot GEMMs.
template <int K>
__device__ __forceinline__ void gemm_mainloop(const bf16_t* arow, const bf16_t* BT, int ldk,
                                              int n0, int lane, v8f c[8]) {
  const bf16_t* brow[8];
#pragma unroll
  for (int j = 0; j < 8; ++j)
    brow[j] = BT + (size_t)(n0 + j * 16 + (lane & 15)) * ldk + ((lane >> 4) << 3);

  v16bf a_cur = load_A_row(arow, 0);
  v16bf b_cur[8];
#pragma unroll
  for (int j = 0; j < 8; ++j)
    b_cur[j] = joinv(*(const v8bf*)(brow[j]), *(const v8bf*)(brow[j] + 16));

  for (int k = 0; k < K - 32; k += 32) {
    int kn = k + 32;
    v16bf a_nxt = load_A_row(arow, kn);
    v16bf b_nxt[8];
#pragma unroll
    for (int j = 0; j < 8; ++j)
      b_nxt[j] = joinv(*(const v8bf*)(brow[j] + kn), *(const v8bf*)(brow[j] + kn + 16));
    if ((k & 127) == 0 && k + 160 < K) {
      __builtin_prefetch((const void*)(arow + k + 160), 0, 3);
#pragma unroll
      for (int j = 0; j < 8; ++j) __builtin_prefetch((const void*)(brow[j] + k + 160), 0, 3);
    }
#pragma unroll
    for (int j = 0; j < 8; ++j) c[j] = wmma_bf16f32(a_cur, b_cur[j], c[j]);
    a_cur = a_nxt;
#pragma unroll
    for (int j = 0; j < 8; ++j) b_cur[j] = b_nxt[j];
  }
#pragma unroll
  for (int j = 0; j < 8; ++j) c[j] = wmma_bf16f32(a_cur, b_cur[j], c[j]);
}

// ---------------- transpose + convert: out[N][K] = (bf16) in[K][N], batched ----------------

__global__ __launch_bounds__(256) void transpose_cvt_kernel(const float* __restrict__ in,
                                                            bf16_t* __restrict__ out,
                                                            int K, int N) {
  __shared__ bf16_t tile[32][33];
  size_t boff = (size_t)blockIdx.z * K * N;
  const float* src = in + boff;
  bf16_t* dst = out + boff;
  int k0 = blockIdx.y * 32;
  int n0 = blockIdx.x * 32;
  int tx = threadIdx.x & 31;
  int ty = threadIdx.x >> 5;
#pragma unroll
  for (int r = ty; r < 32; r += 8)
    tile[r][tx] = (bf16_t)src[(size_t)(k0 + r) * N + n0 + tx];
  __syncthreads();
#pragma unroll
  for (int r = ty; r < 32; r += 8)
    dst[(size_t)(n0 + r) * K + k0 + tx] = tile[tx][r];
}

// per-head V transpose: vt[bh][hd][s] = v[(b*S+s)*D + h*64 + hd]
__global__ __launch_bounds__(256) void transpose_v_kernel(const bf16_t* __restrict__ vb,
                                                          bf16_t* __restrict__ vt) {
  __shared__ bf16_t tile[32][72];
  int bh = blockIdx.y;
  int b = bh >> 4, h = bh & 15;
  int s0 = blockIdx.x * 32;
  int tx = threadIdx.x & 31;
  int ty = threadIdx.x >> 5;
#pragma unroll
  for (int r = ty; r < 32; r += 8) {
    const bf16_t* src = vb + (size_t)(b * SEQ + s0 + r) * DMODEL + h * HDIM;
    tile[r][tx] = src[tx];
    tile[r][tx + 32] = src[tx + 32];
  }
  __syncthreads();
  bf16_t* dst = vt + (size_t)bh * HDIM * SEQ;
#pragma unroll
  for (int hd = ty; hd < HDIM; hd += 8)
    dst[(size_t)hd * SEQ + s0 + tx] = tile[tx][hd];
}

// ---------------- norm ----------------

__global__ __launch_bounds__(256) void rmsnorm_kernel(const float* __restrict__ x,
                                                      const float* __restrict__ w,
                                                      bf16_t* __restrict__ outb,
                                                      float* __restrict__ outf) {
  __shared__ float red[8];
  __shared__ float rinv_s;
  int t = blockIdx.x;
  const float* row = x + (size_t)t * DMODEL;
  float ss = 0.f;
  for (int d = threadIdx.x; d < DMODEL; d += 256) {
    float v = row[d];
    ss += v * v;
  }
#pragma unroll
  for (int m = 16; m; m >>= 1) ss += __shfl_xor(ss, m, 32);
  if ((threadIdx.x & 31) == 0) red[threadIdx.x >> 5] = ss;
  __syncthreads();
  if (threadIdx.x == 0) {
    float s = 0.f;
#pragma unroll
    for (int i = 0; i < 8; ++i) s += red[i];
    rinv_s = rsqrtf(s / (float)DMODEL + 1e-5f);
  }
  __syncthreads();
  float rv = rinv_s;
  for (int d = threadIdx.x; d < DMODEL; d += 256) {
    float v = row[d] * rv * w[d];
    outb[(size_t)t * DMODEL + d] = (bf16_t)v;
    if (outf) outf[(size_t)t * DMODEL + d] = v;
  }
}

// ---------------- bf16 WMMA GEMM with transposed B: C[M,N] = A[M,K] @ BT[N,K]^T ----------------

template <int N, int K>
__global__ __launch_bounds__(256) void gemm_bf16_kernel(const bf16_t* __restrict__ A,
                                                        const bf16_t* __restrict__ BT,
                                                        const float* __restrict__ resid,
                                                        float* __restrict__ outF,
                                                        bf16_t* __restrict__ outB) {
  int lane = threadIdx.x & 31, wave = threadIdx.x >> 5;
  int m0 = blockIdx.y * 128 + wave * 16;
  int n0 = blockIdx.x * 128;
  const bf16_t* arow = A + (size_t)(m0 + (lane & 15)) * K + ((lane >> 4) << 3);
  v8f c[8] = {};
  gemm_mainloop<K>(arow, BT, K, n0, lane, c);

  int r0 = m0 + ((lane >> 4) << 3);
  int cc = n0 + (lane & 15);
#pragma unroll
  for (int j = 0; j < 8; ++j)
#pragma unroll
    for (int i = 0; i < 8; ++i) {
      size_t idx = (size_t)(r0 + i) * N + cc + j * 16;
      float v = c[j][i];
      if (outB)
        outB[idx] = (bf16_t)v;
      else
        outF[idx] = (resid ? resid[idx] : 0.f) + v;
    }
}

// ---------------- flash attention (streaming softmax, WMMA) ----------------

__global__ __launch_bounds__(128) void attn_kernel(const bf16_t* __restrict__ qb,
                                                   const bf16_t* __restrict__ kb,
                                                   const bf16_t* __restrict__ vt,
                                                   bf16_t* __restrict__ ctxb) {
  __shared__ bf16_t plds[4][16][72];
  int lane = threadIdx.x & 31, wave = threadIdx.x >> 5;
  int b = blockIdx.y >> 4, h = blockIdx.y & 15;
  int q0 = blockIdx.x * 64 + wave * 16;
  size_t headoff = (size_t)(b * SEQ) * DMODEL + h * HDIM;
  const bf16_t* qbase = qb + headoff;
  const bf16_t* kbase = kb + headoff;
  const bf16_t* vtbase = vt + (size_t)blockIdx.y * HDIM * SEQ;  // [hd][s]

  v16bf qa0 = load_A_frag(qbase, DMODEL, q0, 0, lane);
  v16bf qa1 = load_A_frag(qbase, DMODEL, q0, 32, lane);

  float mrow[8], lrow[8];
  v8f acc[4] = {};
#pragma unroll
  for (int i = 0; i < 8; ++i) {
    mrow[i] = -1e30f;
    lrow[i] = 0.f;
  }

  for (int key0 = 0; key0 < SEQ; key0 += 64) {
    v8f s[4] = {};
#pragma unroll
    for (int j = 0; j < 4; ++j) {
      v16bf bk0 = load_BT_frag(kbase, DMODEL, key0 + j * 16, 0, lane);
      s[j] = wmma_bf16f32(qa0, bk0, s[j]);
      v16bf bk1 = load_BT_frag(kbase, DMODEL, key0 + j * 16, 32, lane);
      s[j] = wmma_bf16f32(qa1, bk1, s[j]);
    }
#pragma unroll
    for (int j = 0; j < 4; ++j)
#pragma unroll
      for (int i = 0; i < 8; ++i) s[j][i] *= 0.125f;  // 1/sqrt(64)

#pragma unroll
    for (int i = 0; i < 8; ++i) {
      float v = fmaxf(fmaxf(s[0][i], s[1][i]), fmaxf(s[2][i], s[3][i]));
#pragma unroll
      for (int m = 1; m < 16; m <<= 1) v = fmaxf(v, __shfl_xor(v, m, 32));
      float mnew = fmaxf(mrow[i], v);
      float alpha = __expf(mrow[i] - mnew);
      mrow[i] = mnew;
#pragma unroll
      for (int j = 0; j < 4; ++j) {
        s[j][i] = __expf(s[j][i] - mnew);
        acc[j][i] *= alpha;
      }
      float sm = s[0][i] + s[1][i] + s[2][i] + s[3][i];
#pragma unroll
      for (int m = 1; m < 16; m <<= 1) sm += __shfl_xor(sm, m, 32);
      lrow[i] = lrow[i] * alpha + sm;
    }

#pragma unroll
    for (int j = 0; j < 4; ++j)
#pragma unroll
      for (int i = 0; i < 8; ++i)
        plds[wave][((lane >> 4) << 3) + i][j * 16 + (lane & 15)] = (bf16_t)s[j][i];
    __syncthreads();
    v16bf pa0 = load_A_lds(&plds[wave][0][0], 72, 0, lane);
    v16bf pa1 = load_A_lds(&plds[wave][0][0], 72, 32, lane);
#pragma unroll
    for (int j = 0; j < 4; ++j) {
      v16bf vb0 = load_BT_frag(vtbase, SEQ, j * 16, key0, lane);
      acc[j] = wmma_bf16f32(pa0, vb0, acc[j]);
      v16bf vb1 = load_BT_frag(vtbase, SEQ, j * 16, key0 + 32, lane);
      acc[j] = wmma_bf16f32(pa1, vb1, acc[j]);
    }
    __syncthreads();
  }

  int r0 = q0 + ((lane >> 4) << 3);
  int c0 = lane & 15;
#pragma unroll
  for (int j = 0; j < 4; ++j)
#pragma unroll
    for (int i = 0; i < 8; ++i)
      ctxb[(size_t)(b * SEQ + r0 + i) * DMODEL + h * HDIM + j * 16 + c0] =
          (bf16_t)(acc[j][i] / lrow[i]);
}

// ---------------- router ----------------

__global__ __launch_bounds__(256) void router_kernel(const float* __restrict__ h2f,
                                                     const float* __restrict__ wr,
                                                     int* __restrict__ cnts,
                                                     int* __restrict__ lists,
                                                     float* __restrict__ topw,
                                                     float* __restrict__ sumprob,
                                                     float* __restrict__ sumzz) {
  int t = blockIdx.x * 8 + (threadIdx.x >> 5);
  int lane = threadIdx.x & 31;
  float acc[NEXP];
#pragma unroll
  for (int e = 0; e < NEXP; ++e) acc[e] = 0.f;
  const float* hrow = h2f + (size_t)t * DMODEL;
  for (int d = lane; d < DMODEL; d += 32) {
    float hv = hrow[d];
    const float* w = wr + (size_t)d * NEXP;
#pragma unroll
    for (int e = 0; e < NEXP; ++e) acc[e] += hv * w[e];
  }
#pragma unroll
  for (int m = 16; m; m >>= 1)
#pragma unroll
    for (int e = 0; e < NEXP; ++e) acc[e] += __shfl_xor(acc[e], m, 32);

  if (lane == 0) {
    float mx = acc[0];
#pragma unroll
    for (int e = 1; e < NEXP; ++e) mx = fmaxf(mx, acc[e]);
    float p[NEXP], se = 0.f;
#pragma unroll
    for (int e = 0; e < NEXP; ++e) {
      p[e] = __expf(acc[e] - mx);
      se += p[e];
    }
    float inv = 1.f / se;
#pragma unroll
    for (int e = 0; e < NEXP; ++e) p[e] *= inv;
    float z = logf(se) + mx;
    atomicAdd(sumzz, z * z);
#pragma unroll
    for (int e = 0; e < NEXP; ++e) atomicAdd(&sumprob[e], p[e]);
    int i1 = 0;
#pragma unroll
    for (int e = 1; e < NEXP; ++e)
      if (p[e] > p[i1]) i1 = e;
    int i2 = (i1 == 0) ? 1 : 0;
#pragma unroll
    for (int e = 0; e < NEXP; ++e)
      if (e != i1 && p[e] > p[i2]) i2 = e;
    float ws = p[i1] + p[i2];
    int pos1 = atomicAdd(&cnts[i1], 1);
    lists[i1 * T_TOK + pos1] = t * 2 + 0;
    topw[t * 2 + 0] = p[i1] / ws;
    int pos2 = atomicAdd(&cnts[i2], 1);
    lists[i2 * T_TOK + pos2] = t * 2 + 1;
    topw[t * 2 + 1] = p[i2] / ws;
  }
}

// ---------------- grouped expert GEMM 1 (TDM-fed): he = gelu(h2 @ w1[e]) ----------------
// grid (F/128, 32 mtiles(128), E), block 256 = 8 waves, wave 16m x 128n

__global__ __launch_bounds__(256) void expert_gemm1_kernel(const bf16_t* __restrict__ h2b,
                                                           const bf16_t* __restrict__ w1T,
                                                           const int* __restrict__ cnts,
                                                           const int* __restrict__ lists,
                                                           bf16_t* __restrict__ heb) {
  __shared__ char sbuf[2 * TDM_BUFB];
  int e = blockIdx.z;
  int cnt = cnts[e];
  int mblk = blockIdx.y * 128;
  if (mblk >= cnt) return;
  const int* lst = lists + e * T_TOK;
  int lane = threadIdx.x & 31, wave = threadIdx.x >> 5;
  int m0 = mblk + wave * 16;
  int n0 = blockIdx.x * 128;
  const bf16_t* Wtile = w1T + (size_t)e * DMODEL * FFN + (size_t)n0 * DMODEL;

  int r = m0 + (lane & 15);
  if (r >= cnt) r = cnt - 1;
  int token = lst[r] >> 1;
  const bf16_t* arow = h2b + (size_t)token * DMODEL + ((lane >> 4) << 3);

  unsigned sbase = (unsigned)(uintptr_t)&sbuf[0];
  v8f c[8] = {};
  gemm_mainloop_tdm<DMODEL>(arow, Wtile, DMODEL, lane, wave, sbuf, sbase, c);

  int rr0 = m0 + ((lane >> 4) << 3);
  int cc = n0 + (lane & 15);
#pragma unroll
  for (int i = 0; i < 8; ++i) {
    int rg = rr0 + i;
    if (rg >= cnt) continue;
    size_t ts = (size_t)lst[rg];
#pragma unroll
    for (int j = 0; j < 8; ++j)
      heb[ts * FFN + cc + j * 16] = (bf16_t)gelu_f(c[j][i]);
  }
}

// ---------------- grouped expert GEMM 2 (TDM-fed): out2 = weight * (he @ w2[e]) ----------------
// grid (D/128, 32 mtiles(128), E)

__global__ __launch_bounds__(256) void expert_gemm2_kernel(const bf16_t* __restrict__ heb,
                                                           const bf16_t* __restrict__ w2T,
                                                           const int* __restrict__ cnts,
                                                           const int* __restrict__ lists,
                                                           const float* __restrict__ topw,
                                                           float* __restrict__ out2) {
  __shared__ char sbuf[2 * TDM_BUFB];
  int e = blockIdx.z;
  int cnt = cnts[e];
  int mblk = blockIdx.y * 128;
  if (mblk >= cnt) return;
  const int* lst = lists + e * T_TOK;
  int lane = threadIdx.x & 31, wave = threadIdx.x >> 5;
  int m0 = mblk + wave * 16;
  int n0 = blockIdx.x * 128;
  const bf16_t* Wtile = w2T + (size_t)e * FFN * DMODEL + (size_t)n0 * FFN;

  int r = m0 + (lane & 15);
  if (r >= cnt) r = cnt - 1;
  size_t tsrow = (size_t)lst[r];
  const bf16_t* arow = heb + tsrow * FFN + ((lane >> 4) << 3);

  unsigned sbase = (unsigned)(uintptr_t)&sbuf[0];
  v8f c[8] = {};
  gemm_mainloop_tdm<FFN>(arow, Wtile, FFN, lane, wave, sbuf, sbase, c);

  int rr0 = m0 + ((lane >> 4) << 3);
  int cc = n0 + (lane & 15);
#pragma unroll
  for (int i = 0; i < 8; ++i) {
    int rg = rr0 + i;
    if (rg >= cnt) continue;
    size_t ts = (size_t)lst[rg];
    float w = topw[ts];
#pragma unroll
    for (int j = 0; j < 8; ++j) out2[ts * DMODEL + cc + j * 16] = w * c[j][i];
  }
}

// ---------------- combine + loss ----------------

__global__ void combine_kernel(const float* __restrict__ x1, const float* __restrict__ out2,
                               float* __restrict__ out) {
  int i = blockIdx.x * blockDim.x + threadIdx.x;
  int t = i >> 10;
  int d = i & 1023;
  out[i] = x1[i] + out2[(size_t)(t * 2) * DMODEL + d] + out2[(size_t)(t * 2 + 1) * DMODEL + d];
}

__global__ void init_stats_kernel(int* cnts, float* sumprob, float* sumzz) {
  if (threadIdx.x < NEXP) {
    cnts[threadIdx.x] = 0;
    sumprob[threadIdx.x] = 0.f;
  }
  if (threadIdx.x == 0) sumzz[0] = 0.f;
}

__global__ void loss_kernel(const int* __restrict__ cnts, const float* __restrict__ sumprob,
                            const float* __restrict__ sumzz, float* __restrict__ out) {
  float zl = 0.001f * (sumzz[0] / (float)T_TOK);
  float s = 0.f;
#pragma unroll
  for (int e = 0; e < NEXP; ++e)
    s += (sumprob[e] / (float)T_TOK) * ((float)cnts[e] / (float)T_TOK);
  out[0] = zl + (float)NEXP * s;
}

// ---------------- host launcher ----------------

extern "C" void kernel_launch(void* const* d_in, const int* in_sizes, int n_in,
                              void* d_out, int out_size, void* d_ws, size_t ws_size,
                              hipStream_t stream) {
  const float* x      = (const float*)d_in[0];
  const float* ln1_w  = (const float*)d_in[1];
  const float* ln2_w  = (const float*)d_in[2];
  const float* wq     = (const float*)d_in[3];
  const float* wk     = (const float*)d_in[4];
  const float* wv     = (const float*)d_in[5];
  const float* wo     = (const float*)d_in[6];
  const float* w_rout = (const float*)d_in[7];
  const float* w1     = (const float*)d_in[8];
  const float* w2     = (const float*)d_in[9];
  float* out = (float*)d_out;

  char* ws = (char*)d_ws;
  size_t off = 0;
  auto alloc = [&](size_t bytes) {
    size_t o = off;
    off += (bytes + 255) & ~(size_t)255;
    return o;
  };
  const size_t TD = (size_t)T_TOK * DMODEL;
  bf16_t* h1b  = (bf16_t*)(ws + alloc(TD * 2));
  bf16_t* qb   = (bf16_t*)(ws + alloc(TD * 2));
  bf16_t* kb   = (bf16_t*)(ws + alloc(TD * 2));
  bf16_t* vb   = (bf16_t*)(ws + alloc(TD * 2));
  bf16_t* vtb  = (bf16_t*)(ws + alloc(TD * 2));
  bf16_t* ctxb = (bf16_t*)(ws + alloc(TD * 2));
  float*  x1   = (float*)(ws + alloc(TD * 4));
  float*  h2f  = (float*)(ws + alloc(TD * 4));
  bf16_t* h2b  = (bf16_t*)(ws + alloc(TD * 2));
  bf16_t* wqT  = (bf16_t*)(ws + alloc((size_t)DMODEL * DMODEL * 2));
  bf16_t* wkT  = (bf16_t*)(ws + alloc((size_t)DMODEL * DMODEL * 2));
  bf16_t* wvT  = (bf16_t*)(ws + alloc((size_t)DMODEL * DMODEL * 2));
  bf16_t* woT  = (bf16_t*)(ws + alloc((size_t)DMODEL * DMODEL * 2));
  bf16_t* w1T  = (bf16_t*)(ws + alloc((size_t)NEXP * DMODEL * FFN * 2));
  bf16_t* w2T  = (bf16_t*)(ws + alloc((size_t)NEXP * FFN * DMODEL * 2));
  bf16_t* heb  = (bf16_t*)(ws + alloc((size_t)T_TOK * 2 * FFN * 2));
  float*  out2 = (float*)(ws + alloc((size_t)T_TOK * 2 * DMODEL * 4));
  int*    lists = (int*)(ws + alloc((size_t)NEXP * T_TOK * 4));
  int*    cnts  = (int*)(ws + alloc(NEXP * 4));
  float*  sumprob = (float*)(ws + alloc(NEXP * 4));
  float*  sumzz   = (float*)(ws + alloc(4));
  float*  topw    = (float*)(ws + alloc((size_t)T_TOK * 2 * 4));

  // 1) transpose-convert all weights to bf16 (WT[N][K])
  hipLaunchKernelGGL(transpose_cvt_kernel, dim3(32, 32, 1), dim3(256), 0, stream, wq, wqT, DMODEL, DMODEL);
  hipLaunchKernelGGL(transpose_cvt_kernel, dim3(32, 32, 1), dim3(256), 0, stream, wk, wkT, DMODEL, DMODEL);
  hipLaunchKernelGGL(transpose_cvt_kernel, dim3(32, 32, 1), dim3(256), 0, stream, wv, wvT, DMODEL, DMODEL);
  hipLaunchKernelGGL(transpose_cvt_kernel, dim3(32, 32, 1), dim3(256), 0, stream, wo, woT, DMODEL, DMODEL);
  hipLaunchKernelGGL(transpose_cvt_kernel, dim3(FFN / 32, DMODEL / 32, NEXP), dim3(256), 0, stream,
                     w1, w1T, DMODEL, FFN);
  hipLaunchKernelGGL(transpose_cvt_kernel, dim3(DMODEL / 32, FFN / 32, NEXP), dim3(256), 0, stream,
                     w2, w2T, FFN, DMODEL);

  // 2) rmsnorm 1
  hipLaunchKernelGGL(rmsnorm_kernel, dim3(T_TOK), dim3(256), 0, stream, x, ln1_w, h1b, (float*)nullptr);

  // 3) q/k/v projections (bf16 out)
  hipLaunchKernelGGL((gemm_bf16_kernel<DMODEL, DMODEL>), dim3(8, 32), dim3(256), 0, stream, h1b, wqT,
                     (const float*)nullptr, (float*)nullptr, qb);
  hipLaunchKernelGGL((gemm_bf16_kernel<DMODEL, DMODEL>), dim3(8, 32), dim3(256), 0, stream, h1b, wkT,
                     (const float*)nullptr, (float*)nullptr, kb);
  hipLaunchKernelGGL((gemm_bf16_kernel<DMODEL, DMODEL>), dim3(8, 32), dim3(256), 0, stream, h1b, wvT,
                     (const float*)nullptr, (float*)nullptr, vb);

  // 4) per-head V transpose + flash attention
  hipLaunchKernelGGL(transpose_v_kernel, dim3(SEQ / 32, 4 * NHEADS), dim3(256), 0, stream, vb, vtb);
  hipLaunchKernelGGL(attn_kernel, dim3(SEQ / 64, 4 * NHEADS), dim3(128), 0, stream, qb, kb, vtb, ctxb);

  // 5) output projection + residual: x1 = x + ctx @ wo
  hipLaunchKernelGGL((gemm_bf16_kernel<DMODEL, DMODEL>), dim3(8, 32), dim3(256), 0, stream, ctxb, woT,
                     x, x1, (bf16_t*)nullptr);

  // 6) rmsnorm 2 (f32 for router, bf16 for experts)
  hipLaunchKernelGGL(rmsnorm_kernel, dim3(T_TOK), dim3(256), 0, stream, x1, ln2_w, h2b, h2f);

  // 7) router
  hipLaunchKernelGGL(init_stats_kernel, dim3(1), dim3(32), 0, stream, cnts, sumprob, sumzz);
  hipLaunchKernelGGL(router_kernel, dim3(T_TOK / 8), dim3(256), 0, stream, h2f, w_rout, cnts, lists,
                     topw, sumprob, sumzz);

  // 8) grouped expert GEMMs (sparse top-2, TDM-fed double-buffered LDS)
  hipLaunchKernelGGL(expert_gemm1_kernel, dim3(FFN / 128, 32, NEXP), dim3(256), 0, stream, h2b, w1T,
                     cnts, lists, heb);
  hipLaunchKernelGGL(expert_gemm2_kernel, dim3(DMODEL / 128, 32, NEXP), dim3(256), 0, stream, heb,
                     w2T, cnts, lists, topw, out2);

  // 9) combine + loss
  hipLaunchKernelGGL(combine_kernel, dim3((T_TOK * DMODEL) / 256), dim3(256), 0, stream, x1, out2, out);
  hipLaunchKernelGGL(loss_kernel, dim3(1), dim3(1), 0, stream, cnts, sumprob, sumzz, out + TD);
}
